// LDGCNNSegmentor_57174604644617
// MI455X (gfx1250) — matrix operand order repaired and checked by reading, not compiled, not run on hardware
//
#include <hip/hip_runtime.h>
#include <math.h>
#include <stdint.h>

// ---------------------------------------------------------------------------
// CDNA5 (gfx1250) LDGCNN forward. wave32, WMMA f16 -> f32 accumulate.
// ---------------------------------------------------------------------------

typedef _Float16 v16h __attribute__((ext_vector_type(16)));
typedef _Float16 v2h  __attribute__((ext_vector_type(2)));
typedef float    v8f  __attribute__((ext_vector_type(8)));

// async global->LDS builtin operand types (signature leaked by clang
// diagnostics: v4i in AS(1) source, v4i in AS(3) dest, imm offset, imm cpol)
typedef int v4i __attribute__((vector_size(16)));
typedef __attribute__((address_space(1))) v4i as1_v4i;
typedef __attribute__((address_space(3))) v4i as3_v4i;

#if defined(__has_builtin)
#if __has_builtin(__builtin_amdgcn_global_load_async_to_lds_b128) && \
    __has_builtin(__builtin_amdgcn_s_wait_asynccnt)
#define HAS_ASYNC_LDS 1
#endif
#endif
#ifndef HAS_ASYNC_LDS
#define HAS_ASYNC_LDS 0
#endif

#define BATCH   8
#define NPTS    2048
#define NPOINTS (BATCH * NPTS)     // 16384
#define CF      326                // total feature channels (x0|x1|x2|x3|x4)
#define FPAD    352                // CF rounded up to 32 (f16 A-operand stride)
#define KNN     30
#define PSTR    32                 // edge rows per point (30 padded to 32)
#define EPAD    (NPOINTS * PSTR)   // 524288 padded edge rows

// -------------------- WMMA fragment loaders (ISA 7.12.2 layouts) -----------
// A 16x32 f16, lane = m (lane&15), half = lane>>4:
//   VGPR v: k = (v>>2)*16 + half*8 + (v&3)*2   (pair k,k+1)
// B 32x16 f16, lane = n (lane&15), half = lane>>4:
//   VGPR v: k = (v>>2)*16 + half*8 + (v&3)*2   (pair k,k+1)
__device__ __forceinline__ v16h load_a_frag(const _Float16* Arow, int k0, int half) {
  v16h a;
#pragma unroll
  for (int v = 0; v < 8; ++v) {
    int ka = ((v >> 2) << 4) + (half << 3) + ((v & 3) << 1);
    v2h p = *(const v2h*)(Arow + k0 + ka);
    a[2 * v] = p[0]; a[2 * v + 1] = p[1];
  }
  return a;
}

__device__ __forceinline__ v16h load_b_frag(const _Float16* Brow, int k0, int half) {
  v16h b;
#pragma unroll
  for (int v = 0; v < 8; ++v) {
    v2h p = *(const v2h*)(Brow + k0 + ((v >> 2) << 4) + (half << 3) + ((v & 3) << 1));
    b[2 * v] = p[0]; b[2 * v + 1] = p[1];
  }
  return b;
}

// -------------------- small utility kernels --------------------------------
__global__ void fill_h16_zero(_Float16* p, int n) {
  int i = blockIdx.x * blockDim.x + threadIdx.x;
  if (i < n) p[i] = (_Float16)0.f;
}
__global__ void fill_f32(float* p, int n, float v) {
  int i = blockIdx.x * blockDim.x + threadIdx.x;
  if (i < n) p[i] = v;
}

// W f32 [Cin, Cout] -> WT f16 [NP, KP] (transposed, zero padded)
__global__ void convert_wt(const float* W, int Cin, int Cout,
                           _Float16* WT, int KP, int NP) {
  int i = blockIdx.x * blockDim.x + threadIdx.x;
  if (i >= NP * KP) return;
  int n = i / KP, k = i - n * KP;
  float v = (n < Cout && k < Cin) ? W[(size_t)k * Cout + n] : 0.f;
  WT[i] = (_Float16)v;
}

// x0 = cat(x, pos) -> feat32[:, 0:6] and feat16[:, 0:6]
__global__ void build_x0(const float* x, const float* pos,
                         float* feat32, _Float16* feat16) {
  int i = blockIdx.x * blockDim.x + threadIdx.x;
  if (i >= NPOINTS * 6) return;
  int p = i / 6, c = i - p * 6;
  float v = (c < 3) ? x[(size_t)p * 3 + c] : pos[(size_t)p * 3 + (c - 3)];
  feat32[(size_t)p * CF + c] = v;
  feat16[(size_t)p * FPAD + c] = (_Float16)v;
}

// -------------------- brute-force kNN (one block per query) ----------------
__global__ void __launch_bounds__(256)
knn_kernel(const float* feat32, int choff, int D, int* idxout) {
  __shared__ float dist[NPTS];
  __shared__ float xi[64];
  __shared__ float rval[256];
  __shared__ int   ridx[256];
  __shared__ int   selbuf[PSTR];

  int p = blockIdx.x;
  int b = p / NPTS, i = p - b * NPTS;
  const float* base = feat32 + (size_t)b * NPTS * CF + choff;

  for (int c = threadIdx.x; c < D; c += 256) xi[c] = base[(size_t)i * CF + c];
  __syncthreads();

  for (int j = threadIdx.x; j < NPTS; j += 256) {
    const float* xj = base + (size_t)j * CF;
    float s = 0.f;
    for (int c = 0; c < D; ++c) { float d = xj[c] - xi[c]; s += d * d; }
    dist[j] = (j == i) ? INFINITY : s;
  }
  __syncthreads();

  for (int sel = 0; sel < KNN; ++sel) {
    float bv = INFINITY; int bi = 0x7fffffff;
    for (int j = threadIdx.x; j < NPTS; j += 256) {
      float v = dist[j];
      if (v < bv) { bv = v; bi = j; }
    }
    rval[threadIdx.x] = bv; ridx[threadIdx.x] = bi;
    __syncthreads();
    for (int s = 128; s > 0; s >>= 1) {
      if (threadIdx.x < s) {
        float v2 = rval[threadIdx.x + s]; int i2 = ridx[threadIdx.x + s];
        float v1 = rval[threadIdx.x];     int i1 = ridx[threadIdx.x];
        if (v2 < v1 || (v2 == v1 && i2 < i1)) { rval[threadIdx.x] = v2; ridx[threadIdx.x] = i2; }
      }
      __syncthreads();
    }
    if (threadIdx.x == 0) {
      int w = ridx[0];
      selbuf[sel] = w;
      dist[w] = INFINITY;
    }
    __syncthreads();
  }
  if (threadIdx.x < PSTR)
    idxout[(size_t)p * PSTR + threadIdx.x] = selbuf[threadIdx.x < KNN ? threadIdx.x : (KNN - 1)];
}

// -------------------- EdgeConv layer 1: gathered WMMA GEMM -----------------
// All of C / KP / HALF are compile-time: after unrolling, every (k,k+1)
// pair statically resolves to {x_i pair | x_j-x_i pair | zero} -> straight-
// line loads, no EXEC-mask divergence in the hot loop.
template <int C, int KP1, int HALF>
__device__ __forceinline__ v8f edge1_accum(const _Float16* fi, const _Float16* fj,
                                           const _Float16* Brow) {
  v8f acc = {};
#pragma unroll
  for (int k0 = 0; k0 < KP1; k0 += 32) {
    v16h a;
#pragma unroll
    for (int v = 0; v < 8; ++v) {
      const int k = k0 + ((v >> 2) << 4) + (HALF << 3) + ((v & 3) << 1);
      v2h p = {};
      if (k < C) {                       // compile-time selection (C even)
        p = *(const v2h*)(fi + k);
      } else if (k < 2 * C) {
        const int c = k - C;
        v2h xi = *(const v2h*)(fi + c);
        v2h xj = *(const v2h*)(fj + c);
        p = xj - xi;
      }
      a[2 * v] = p[0]; a[2 * v + 1] = p[1];
    }
    v16h bf = load_b_frag(Brow, k0, HALF);
    acc = __builtin_amdgcn_wmma_f32_16x16x32_f16(false, a, false, bf,
                                                 (short)0, acc, false, false);
  }
  return acc;
}

template <int C, int KP1, int H>
__global__ void __launch_bounds__(256)
edge_gemm1(const _Float16* feat16, const int* idx,
           const _Float16* WT, const float* bias, _Float16* hidden) {
  int wave = threadIdx.x >> 5, lane = threadIdx.x & 31;
  int half = lane >> 4, l = lane & 15;
  int mtile = blockIdx.x * 8 + wave;
  int n0 = blockIdx.y * 16;
  int m = mtile * 16 + l;            // edge row this lane gathers
  int p = m >> 5;                    // point
  int b = p / NPTS;
  int kk = m & (PSTR - 1);
  int j = idx[(size_t)p * PSTR + kk];
  const _Float16* fi = feat16 + (size_t)p * FPAD;
  const _Float16* fj = feat16 + ((size_t)b * NPTS + j) * FPAD;
  const _Float16* Brow = WT + (size_t)(n0 + l) * KP1;

  v8f acc = half ? edge1_accum<C, KP1, 1>(fi, fj, Brow)
                 : edge1_accum<C, KP1, 0>(fi, fj, Brow);

#pragma unroll
  for (int r = 0; r < 8; ++r) {
    int mr = mtile * 16 + r + half * 8;
    float v = fmaxf(acc[r] + bias[n0 + l], 0.f);
    hidden[(size_t)mr * H + n0 + l] = (_Float16)v;
  }
}

// -------------------- EdgeConv layer 2 + max over neighbors ----------------
// out[p, :] = max_{kk<30} ( hidden[p*32+kk, :] @ W2 + b2 ), written into feat.
// For Cout=64 the two M-tiles (neighbor halves) are split across wave pairs
// so all 8 waves run WMMA; per-(nt,mt) LDS slots are combined after a barrier.
template <int H, int Cout>
__global__ void __launch_bounds__(256)
edge_gemm2_max(const _Float16* hidden, const _Float16* WT, const float* bias,
               float* feat32, _Float16* feat16, int choff) {
  constexpr int NT  = Cout >> 4;            // n-tiles: 4 or 8
  constexpr int WPM = (NT <= 4) ? 2 : 1;    // waves per n-tile (mt split)
  __shared__ float slot[2][Cout];
  int p = blockIdx.x;
  int wave = threadIdx.x >> 5, lane = threadIdx.x & 31;
  int half = lane >> 4, l = lane & 15;
  int nt = wave % NT;
  int ms = wave / NT;

  if (ms < WPM) {
    const _Float16* Brow = WT + (size_t)(nt * 16 + l) * H;
    float vmax = -INFINITY;
    for (int mt = ms; mt < 2; mt += WPM) {
      const _Float16* Arow = hidden + (size_t)(p * PSTR + mt * 16 + l) * H;
      v8f acc = {};
#pragma unroll
      for (int k0 = 0; k0 < H; k0 += 32) {
        v16h a = load_a_frag(Arow, k0, half);
        v16h bf = load_b_frag(Brow, k0, half);
        acc = __builtin_amdgcn_wmma_f32_16x16x32_f16(false, a, false, bf,
                                                     (short)0, acc, false, false);
      }
#pragma unroll
      for (int r = 0; r < 8; ++r) {
        int kk = mt * 16 + r + half * 8;
        if (kk < KNN) vmax = fmaxf(vmax, acc[r]);
      }
    }
    vmax = fmaxf(vmax, __shfl_xor(vmax, 16, 32));  // combine wave halves
    if (half == 0) slot[ms][nt * 16 + l] = vmax;
  }
  __syncthreads();
  for (int c = threadIdx.x; c < Cout; c += 256) {
    float v = slot[0][c];
    if (WPM == 2) v = fmaxf(v, slot[1][c]);
    v += bias[c];
    feat32[(size_t)p * CF + choff + c] = v;
    feat16[(size_t)p * FPAD + choff + c] = (_Float16)v;
  }
}

// -------------------- generic dense WMMA GEMM (B panel in LDS) -------------
// The 16 x KP B panel for this block's n-tile is staged once into LDS
// (async global->LDS on CDNA5, ASYNCcnt-tracked) and shared by all 8
// M-tiles; B fragments then come from ds_load_b128.
// EPI: 0 = bias+relu -> f16, 2 = bias -> f32 (col<Nreal),
//      3 = bias then global atomic column-max into gmax (no store).
template <int EPI>
__global__ void __launch_bounds__(256)
gemm_f16(const _Float16* A, int lda, int M, const _Float16* WT, int KP,
         int Nreal, const float* bias, _Float16* out16, float* out32,
         int ldo, float* gmax) {
  extern __shared__ unsigned char smem_raw[];
  _Float16* Bt = (_Float16*)smem_raw;            // [16][KP]
  int wave = threadIdx.x >> 5, lane = threadIdx.x & 31;
  int half = lane >> 4, l = lane & 15;
  int n0 = blockIdx.y * 16;

  {  // cooperative stage of the B panel (2*KP 16-byte chunks)
    const uint32_t* Bg = (const uint32_t*)(WT + (size_t)n0 * KP);
    uint32_t* Bl = (uint32_t*)Bt;
    int total4 = 2 * KP;
#if HAS_ASYNC_LDS
    for (int d = threadIdx.x; d < total4; d += 256)
      __builtin_amdgcn_global_load_async_to_lds_b128(
          (as1_v4i*)(Bg + 4 * (size_t)d), (as3_v4i*)(Bl + 4 * (size_t)d), 0, 0);
    __builtin_amdgcn_s_wait_asynccnt(0);
#else
    for (int d = threadIdx.x; d < total4; d += 256)
      ((uint4*)Bl)[d] = ((const uint4*)Bg)[d];
#endif
  }
  __syncthreads();

  int mtile = blockIdx.x * 8 + wave;
  int m0 = mtile * 16;
  if (m0 >= M) return;

  const _Float16* Arow = A + (size_t)(m0 + l) * lda;
  const _Float16* Brow = Bt + (size_t)l * KP;

  v8f acc = {};
  for (int k0 = 0; k0 < KP; k0 += 32) {
    __builtin_prefetch(Arow + k0 + 64, 0, 1);
    v16h a = load_a_frag(Arow, k0, half);
    v16h bf = load_b_frag(Brow, k0, half);
    acc = __builtin_amdgcn_wmma_f32_16x16x32_f16(false, a, false, bf,
                                                 (short)0, acc, false, false);
  }

  int n = n0 + l;
  float vb = (n < Nreal) ? bias[n] : 0.f;
  if (EPI == 3) {
    float vmax = -INFINITY;
#pragma unroll
    for (int r = 0; r < 8; ++r) vmax = fmaxf(vmax, acc[r]);
    vmax = fmaxf(vmax, __shfl_xor(vmax, 16, 32));
    if (half == 0) atomicMax(gmax + n, vmax + vb);   // global_atomic_max_num_f32
  } else {
#pragma unroll
    for (int r = 0; r < 8; ++r) {
      int mr = m0 + r + half * 8;
      float v = acc[r] + vb;
      if (EPI == 0) out16[(size_t)mr * ldo + n] = (_Float16)fmaxf(v, 0.f);
      if (EPI == 2) { if (n < Nreal) out32[(size_t)mr * ldo + n] = v; }
    }
  }
}

// -------------------- head input: [link4(326) | g(1024) | pad] -------------
__global__ void build_hin(const _Float16* feat16, const float* g,
                          _Float16* hin, int HKP) {
  int i = blockIdx.x * blockDim.x + threadIdx.x;
  if (i >= NPOINTS * 1376) return;
  int p = i / 1376, c = i - p * 1376;
  _Float16 v;
  if (c < CF)             v = feat16[(size_t)p * FPAD + c];
  else if (c < CF + 1024) v = (_Float16)g[c - CF];
  else                    v = (_Float16)0.f;
  hin[(size_t)p * HKP + c] = v;
}

// -------------------- log_softmax over 50 classes (1 wave / row) -----------
__global__ void __launch_bounds__(32)
logsoftmax50(float* out) {
  int row = blockIdx.x, c = threadIdx.x;
  float* o = out + (size_t)row * 50;
  float v0 = (c < 50) ? o[c] : -INFINITY;
  float v1 = (c + 32 < 50) ? o[c + 32] : -INFINITY;
  float m = fmaxf(v0, v1);
#pragma unroll
  for (int s = 16; s > 0; s >>= 1) m = fmaxf(m, __shfl_xor(m, s, 32));
  float e = ((c < 50) ? expf(v0 - m) : 0.f) + ((c + 32 < 50) ? expf(v1 - m) : 0.f);
#pragma unroll
  for (int s = 16; s > 0; s >>= 1) e += __shfl_xor(e, s, 32);
  float lse = logf(e) + m;
  if (c < 50) o[c] = v0 - lse;
  if (c + 32 < 50) o[c + 32] = v1 - lse;
}

// ===========================================================================
extern "C" void kernel_launch(void* const* d_in, const int* in_sizes, int n_in,
                              void* d_out, int out_size, void* d_ws, size_t ws_size,
                              hipStream_t stream) {
  (void)in_sizes; (void)n_in; (void)out_size; (void)ws_size;
  const float* x   = (const float*)d_in[0];
  const float* pos = (const float*)d_in[1];
  // params flat: ec1(W,b,W,b) ec2 ec3 ec4 fe head(W,b x4)
  const float* P[28];
  for (int i = 0; i < 28; ++i) P[i] = (const float*)d_in[2 + i];

  // ---- workspace carve-out ----
  char* w = (char*)d_ws;
  auto alloc = [&](size_t bytes) -> char* {
    char* r = w; w += (bytes + 255) & ~(size_t)255; return r;
  };
  float*    feat32 = (float*)   alloc((size_t)NPOINTS * CF * 4);
  _Float16* feat16 = (_Float16*)alloc((size_t)NPOINTS * FPAD * 2);
  int*      idx    = (int*)     alloc((size_t)NPOINTS * PSTR * 4);
  _Float16* hidden = (_Float16*)alloc((size_t)EPAD * 128 * 2);   // reused as fe-h1
  float*    g      = (float*)   alloc(1024 * 4);
  _Float16* hin    = (_Float16*)alloc((size_t)NPOINTS * 1376 * 2);
  _Float16* t1     = (_Float16*)alloc((size_t)NPOINTS * 256 * 2);
  _Float16* t2     = (_Float16*)alloc((size_t)NPOINTS * 256 * 2);
  _Float16* t3     = (_Float16*)alloc((size_t)NPOINTS * 128 * 2);

  // converted weights (transposed f16, K padded to 32)
  const int LCI[14] = {12, 64, 140, 64, 268, 64, 396, 128, 326, 1024, 1350, 256, 256, 128};
  const int LCO[14] = {64, 64,  64, 64,  64, 64, 128, 128, 1024, 1024, 256, 256, 128,  50};
  _Float16* WT[14]; int KP[14], NP[14];
  for (int i = 0; i < 14; ++i) {
    KP[i] = (LCI[i] + 31) & ~31;
    NP[i] = (LCO[i] + 15) & ~15;
    WT[i] = (_Float16*)alloc((size_t)NP[i] * KP[i] * 2);
  }

  const int THR = 256;
  auto blks = [](size_t n, int t) { return (unsigned)((n + t - 1) / t); };

  // zero feat16 (covers K-pad channels), convert weights
  fill_h16_zero<<<blks((size_t)NPOINTS * FPAD, THR), THR, 0, stream>>>(feat16, NPOINTS * FPAD);
  for (int i = 0; i < 14; ++i)
    convert_wt<<<blks((size_t)NP[i] * KP[i], THR), THR, 0, stream>>>(
        P[2 * i], LCI[i], LCO[i], WT[i], KP[i], NP[i]);

  build_x0<<<blks((size_t)NPOINTS * 6, THR), THR, 0, stream>>>(x, pos, feat32, feat16);

  const dim3 eg1_64(EPAD / (16 * 8), 64 / 16);
  const dim3 eg1_128(EPAD / (16 * 8), 128 / 16);

  // ---- EdgeConv 1: knn(x0, D=6), MLP 12->64->64 ----
  knn_kernel<<<NPOINTS, THR, 0, stream>>>(feat32, 0, 6, idx);
  edge_gemm1<6, 32, 64><<<eg1_64, THR, 0, stream>>>(feat16, idx, WT[0], P[1], hidden);
  edge_gemm2_max<64, 64><<<NPOINTS, THR, 0, stream>>>(hidden, WT[1], P[3], feat32, feat16, 6);

  // ---- EdgeConv 2: knn(x1, D=64), MLP 140->64->64 ----
  knn_kernel<<<NPOINTS, THR, 0, stream>>>(feat32, 6, 64, idx);
  edge_gemm1<70, 160, 64><<<eg1_64, THR, 0, stream>>>(feat16, idx, WT[2], P[5], hidden);
  edge_gemm2_max<64, 64><<<NPOINTS, THR, 0, stream>>>(hidden, WT[3], P[7], feat32, feat16, 70);

  // ---- EdgeConv 3: knn(x2, D=64), MLP 268->64->64 ----
  knn_kernel<<<NPOINTS, THR, 0, stream>>>(feat32, 70, 64, idx);
  edge_gemm1<134, 288, 64><<<eg1_64, THR, 0, stream>>>(feat16, idx, WT[4], P[9], hidden);
  edge_gemm2_max<64, 64><<<NPOINTS, THR, 0, stream>>>(hidden, WT[5], P[11], feat32, feat16, 134);

  // ---- EdgeConv 4: reuse knn(x2), MLP 396->128->128 ----
  edge_gemm1<198, 416, 128><<<eg1_128, THR, 0, stream>>>(feat16, idx, WT[6], P[13], hidden);
  edge_gemm2_max<128, 128><<<NPOINTS, THR, 0, stream>>>(hidden, WT[7], P[15], feat32, feat16, 198);

  // ---- feature extractor: 326 -> 1024 (relu) -> 1024, fused global max ----
  _Float16* h1 = hidden;  // reuse (16384*1024 f16 fits)
  gemm_f16<0><<<dim3(NPOINTS / 128, 1024 / 16), THR, (size_t)KP[8] * 32, stream>>>(
      feat16, FPAD, NPOINTS, WT[8], KP[8], 1024, P[17], h1, nullptr, 1024, nullptr);
  fill_f32<<<blks(1024, THR), THR, 0, stream>>>(g, 1024, -INFINITY);
  gemm_f16<3><<<dim3(NPOINTS / 128, 1024 / 16), THR, (size_t)KP[9] * 32, stream>>>(
      h1, 1024, NPOINTS, WT[9], KP[9], 1024, P[19], nullptr, nullptr, 0, g);

  // ---- head: cat(link4, g) 1350 -> 256 -> 256 -> 128 -> 50, log_softmax ---
  build_hin<<<blks((size_t)NPOINTS * 1376, THR), THR, 0, stream>>>(feat16, g, hin, KP[10]);
  gemm_f16<0><<<dim3(NPOINTS / 128, 256 / 16), THR, (size_t)KP[10] * 32, stream>>>(
      hin, KP[10], NPOINTS, WT[10], KP[10], 256, P[21], t1, nullptr, 256, nullptr);
  gemm_f16<0><<<dim3(NPOINTS / 128, 256 / 16), THR, (size_t)KP[11] * 32, stream>>>(
      t1, 256, NPOINTS, WT[11], KP[11], 256, P[23], t2, nullptr, 256, nullptr);
  gemm_f16<0><<<dim3(NPOINTS / 128, 128 / 16), THR, (size_t)KP[12] * 32, stream>>>(
      t2, 256, NPOINTS, WT[12], KP[12], 128, P[25], t3, nullptr, 128, nullptr);
  gemm_f16<2><<<dim3(NPOINTS / 128, 64 / 16), THR, (size_t)KP[13] * 32, stream>>>(
      t3, 128, NPOINTS, WT[13], KP[13], 50, P[27], nullptr, (float*)d_out, 50, nullptr);
  logsoftmax50<<<NPOINTS, 32, 0, stream>>>((float*)d_out);
}
